// Discriminator_17377437680454
// MI455X (gfx1250) — matrix operand order repaired and checked by reading, not compiled, not run on hardware
//
#include <hip/hip_runtime.h>
#include <hip/hip_bf16.h>

// ---------------------------------------------------------------------------
// Dimensions (fixed by the reference)
// ---------------------------------------------------------------------------
#define BATCH   1024
#define FEAT    512
#define HIDDEN  1024
#define PREV    10
#define PRED    20
#define TSTEPS  (PREV + PRED)     // 30
#define KCAT    (HIDDEN + FEAT)   // 1536
#define GATES4  (4 * HIDDEN)      // 4096
#define M2      (2 * BATCH)       // 2048 rows: [real batch ; fake batch]

// GEMM tiling for the fused LSTM step
#define KC      32                // K per WMMA chunk (bf16 16x16x32)
#define NITER   (KCAT / KC)       // 48
#define BM      128               // block rows   (2 M-waves x 64)
#define BN      64                // block cols   (4 N-waves x 16)
#define ASTR    80                // LDS row stride (bytes): 64B data + 16B TDM pad
#define BSTR    80
#define ABUF_BYTES (BM * ASTR)            // 10240
#define BBUF_BYTES (4 * BN * BSTR)        // 20480 (4 gates x 64 cols)
#define SMEM_BYTES (2 * ABUF_BYTES + 2 * BBUF_BYTES)   // 61440 < 64KB

typedef __bf16 bf16_t;
typedef __attribute__((ext_vector_type(16))) __bf16        v16bf;
typedef __attribute__((ext_vector_type(8)))  float         v8f;
typedef __attribute__((ext_vector_type(8)))  unsigned int  v8u;
typedef __attribute__((ext_vector_type(4)))  unsigned int  v4u;
typedef __attribute__((ext_vector_type(8)))  int           v8i;
typedef __attribute__((ext_vector_type(4)))  int           v4i;

// ---------------------------------------------------------------------------
// Helpers
// ---------------------------------------------------------------------------
__device__ __forceinline__ unsigned short f2bf_rne(float f) {
    unsigned int u = __float_as_uint(f);
    u += 0x7fffu + ((u >> 16) & 1u);
    return (unsigned short)(u >> 16);
}

__device__ __forceinline__ float sigmoidf_(float x) {
    return 1.0f / (1.0f + __expf(-x));
}

// Build a 16-element bf16 WMMA operand from two 16-byte chunks.
__device__ __forceinline__ v16bf pack_bf16x16(uint4 a, uint4 b) {
    v8u u;
    u[0] = a.x; u[1] = a.y; u[2] = a.z; u[3] = a.w;
    u[4] = b.x; u[5] = b.y; u[6] = b.z; u[7] = b.w;
    return __builtin_bit_cast(v16bf, u);
}

// ---------------------------------------------------------------------------
// TDM descriptor builder + issue (CDNA5 ISA).
//   Up to 3D tile (z planes of y rows of x elements), 2-byte elements,
//   LDS padding: 16B after every 64B tile-row -> 80B LDS row stride.
//   This toolchain exposes the 6-arg builtin:
//   (uint32x4 g0, int32x8 g1, int32x4 g2, int32x4 g3, int32x8 g4, i32 cpol).
// ---------------------------------------------------------------------------
__device__ __forceinline__ void tdm_load(unsigned lds_byte_addr,
                                         const unsigned short* gptr,
                                         unsigned dim0, unsigned dim1, unsigned dim2,
                                         unsigned long long stride_row,    // elems
                                         unsigned long long stride_plane,  // elems
                                         unsigned tile0, unsigned tile1, unsigned tile2)
{
    const unsigned long long ga = (unsigned long long)(uintptr_t)gptr;
    v4u g0;
    g0[0] = 1u;                                  // count=1, user mode, no gather
    g0[1] = lds_byte_addr;                       // D#.lds_addr
    g0[2] = (unsigned)(ga & 0xffffffffu);        // global_addr[31:0]
    g0[3] = (unsigned)((ga >> 32) & 0x01ffffffu) // global_addr[56:32]
          | 0x80000000u;                         // type=2 ("image")
    v8i g1;
    g1[0] = (int)(0x00010000u                    // data_size=1 (2 bytes), wg_mask=0
          | (1u << 20)                           // pad_enable
          | (3u << 22)                           // pad_interval: 16 DWORDs (64B)
          | (3u << 25));                         // pad_amount: 4 DWORDs (16B)
    g1[1] = (int)((dim0 & 0xffffu) << 16);                               // tensor_dim0 lo
    g1[2] = (int)(((dim0 >> 16) & 0xffffu) | ((dim1 & 0xffffu) << 16));  // dim0 hi | dim1 lo
    g1[3] = (int)(((dim1 >> 16) & 0xffffu) | (tile0 << 16));             // dim1 hi | tile_dim0
    g1[4] = (int)((tile1 & 0xffffu) | (tile2 << 16));                    // tile_dim1 | tile_dim2
    g1[5] = (int)(stride_row & 0xffffffffu);                             // dim0_stride lo
    g1[6] = (int)(((stride_row >> 32) & 0xffffu)
          | ((unsigned)(stride_plane & 0xffffu) << 16));                 // hi | dim1_stride lo
    g1[7] = (int)((stride_plane >> 16) & 0xffffffffu);                   // dim1_stride[47:16]
    v4i g2;
    g2[0] = (int)dim2;   // tensor_dim2
    g2[1] = 0;           // tensor_dim3 (iterate off)
    g2[2] = 0;           // tensor_dim2_stride lo (no 4th dim)
    g2[3] = 0;           // stride hi | tile_dim3=0
    v4i g3 = (v4i){0, 0, 0, 0};
    v8i g4 = (v8i){0, 0, 0, 0, 0, 0, 0, 0};      // extra group (unused, zero-filled)
    __builtin_amdgcn_tensor_load_to_lds(g0, g1, g2, g3, g4, 0);
}

// Issue both tiles (A then B) for K-chunk kt into LDS buffer `buf`.
__device__ __forceinline__ void tdm_stage(unsigned lds_base, int kt, int buf,
        const unsigned short* __restrict__ hin,
        const unsigned short* __restrict__ x0,
        const unsigned short* __restrict__ x1,
        const unsigned short* __restrict__ wcat,
        int bx, int by)
{
    // A tile: BM rows x KC elems (2D). Chunk is entirely in h or x; rows of a
    // block are entirely real (bx<8) or fake (bx>=8).
    {
        const unsigned short* gp;
        unsigned d0, d1;
        unsigned long long stride;
        if (kt < HIDDEN) {
            gp = hin + (size_t)bx * BM * HIDDEN + kt;
            d0 = HIDDEN; d1 = M2; stride = HIDDEN;
        } else {
            const int rowG0 = bx * BM;
            gp = (rowG0 < BATCH)
                ? (x0 + (size_t)rowG0 * FEAT + (kt - HIDDEN))
                : (x1 + (size_t)(rowG0 - BATCH) * FEAT + (kt - HIDDEN));
            d0 = FEAT; d1 = BATCH; stride = FEAT;
        }
        tdm_load(lds_base + buf * ABUF_BYTES, gp, d0, d1, 0, stride, 0, KC, BM, 0);
    }
    // B tile: 3D -> 4 gate planes (z-stride HIDDEN*KCAT) x BN rows x KC elems.
    // TDM Z->Y->X walk lands rows in LDS as r = g*BN + col, 80B stride.
    {
        const unsigned short* gp = wcat + (size_t)by * BN * KCAT + kt;
        tdm_load(lds_base + 2 * ABUF_BYTES + buf * BBUF_BYTES, gp,
                 KCAT, HIDDEN, 4, KCAT, (unsigned long long)HIDDEN * KCAT,
                 KC, BN, 4);
    }
}

// ---------------------------------------------------------------------------
// Kernel 1: build WcatT (bf16) = transpose of [Wh ; Wx], shape [4096][1536].
// ---------------------------------------------------------------------------
__global__ void k_build_wcat(const float* __restrict__ Wh,   // [1024][4096]
                             const float* __restrict__ Wx,   // [512][4096]
                             unsigned short* __restrict__ wcat) {
    int tid = blockIdx.x * blockDim.x + threadIdx.x;   // 4096*1536 total
    int n = tid / KCAT;
    int k = tid % KCAT;
    float v = (k < HIDDEN) ? Wh[(size_t)k * GATES4 + n]
                           : Wx[(size_t)(k - HIDDEN) * GATES4 + n];
    wcat[(size_t)n * KCAT + k] = f2bf_rne(v);
}

// ---------------------------------------------------------------------------
// Kernel 2: transpose inputs [B][T][F] f32 -> [T][B][F] bf16
// ---------------------------------------------------------------------------
__global__ void k_transpose_in(const float* __restrict__ src,
                               unsigned short* __restrict__ dst, int T) {
    int tid = blockIdx.x * blockDim.x + threadIdx.x;   // B*T*F total
    int t = tid / (BATCH * FEAT);
    int r = tid % (BATCH * FEAT);
    int bidx = r / FEAT;
    int f = r % FEAT;
    dst[(size_t)t * BATCH * FEAT + (size_t)bidx * FEAT + f] =
        f2bf_rne(src[(size_t)bidx * T * FEAT + (size_t)t * FEAT + f]);
}

// ---------------------------------------------------------------------------
// Kernel 3: fused LSTM step. TDM (async tensor DMA) stages A/B chunks into
//   double-buffered LDS; 8 waves consume via ds_load_b128 + WMMA.
//   Block: 256 threads = 8 waves. Tile: BM=128 rows x BN=64 cols x 4 gates.
//   Wave (mi = w>>2, ni = w&3) owns 64 rows x 16 cols x 4 gates
//   -> 16 accumulators, 16 WMMAs per 32-wide K chunk.
//   Grid = (2048/128, 1024/64) = (16, 16).
// ---------------------------------------------------------------------------
__global__ void __launch_bounds__(256)
k_lstm_step(const unsigned short* __restrict__ hin,   // [2048][1024] bf16
            const unsigned short* __restrict__ x0,    // [1024][512]  bf16 (rows<1024)
            const unsigned short* __restrict__ x1,    // [1024][512]  bf16 (rows>=1024)
            const unsigned short* __restrict__ wcat,  // [4096][1536] bf16
            const float* __restrict__ bias,           // [4096]
            float* __restrict__ cst,                  // [2048][1024] f32 (in/out)
            unsigned short* __restrict__ hout_bf,     // [2048][1024] bf16
            float* __restrict__ hout_f32)             // [2048][1024] f32
{
    __shared__ alignas(16) unsigned char smem[SMEM_BYTES];

    const int tid  = threadIdx.x;
    const int lane = tid & 31;
    const int wave = tid >> 5;
    const int ni = wave & 3;         // 4 waves along N
    const int mi = wave >> 2;        // 2 waves along M (64 rows each)
    const int lm = lane & 15;        // A row / B-C col within 16-wide tile
    const int lh = lane >> 4;        // half selector (ISA §7.12.2 layouts)

    const int bx = blockIdx.x, by = blockIdx.y;
    // Flat LDS aperture address: low 32 bits == LDS byte offset.
    const unsigned lds_base = (unsigned)(uintptr_t)(&smem[0]);

    v8f acc[4][4];                   // [m-tile][gate]
#pragma unroll
    for (int mt = 0; mt < 4; ++mt)
#pragma unroll
        for (int g = 0; g < 4; ++g)
            acc[mt][g] = (v8f){0.f,0.f,0.f,0.f,0.f,0.f,0.f,0.f};

    // Prologue: DMA chunk 0 into buffer 0.
    if (wave == 0) {
        tdm_stage(lds_base, 0, 0, hin, x0, x1, wcat, bx, by);
        __builtin_amdgcn_s_wait_tensorcnt(0);
    }
    __syncthreads();

    for (int i = 0; i < NITER; ++i) {
        const bool havenext = (i + 1 < NITER);
        // Issue next chunk's DMA before computing; it overlaps the WMMAs.
        if (wave == 0 && havenext)
            tdm_stage(lds_base, (i + 1) * KC, (i + 1) & 1, hin, x0, x1, wcat, bx, by);

        const unsigned char* abase = smem + (i & 1) * ABUF_BYTES;
        const unsigned char* bbase = smem + 2 * ABUF_BYTES + (i & 1) * BBUF_BYTES;

        v16bf Av[4], Bv[4];
#pragma unroll
        for (int mt = 0; mt < 4; ++mt) {
            const unsigned char* p = abase + (mi * 64 + mt * 16 + lm) * ASTR;
            // lane half lh: K = {lh*8 .. +7} and {16+lh*8 .. +7}
            Av[mt] = pack_bf16x16(*reinterpret_cast<const uint4*>(p + lh * 16),
                                  *reinterpret_cast<const uint4*>(p + 32 + lh * 16));
        }
#pragma unroll
        for (int g = 0; g < 4; ++g) {
            const unsigned char* p =
                bbase + (g * BN + ni * 16 + lm) * BSTR + lh * 16;
            // lane half lh: K = {lh*8 .. +7} and {16+lh*8 .. +7}
            Bv[g] = pack_bf16x16(*reinterpret_cast<const uint4*>(p),
                                 *reinterpret_cast<const uint4*>(p + 32));
        }
#pragma unroll
        for (int mt = 0; mt < 4; ++mt)
#pragma unroll
            for (int g = 0; g < 4; ++g)
                acc[mt][g] = __builtin_amdgcn_wmma_f32_16x16x32_bf16(
                    false, Av[mt], false, Bv[g], (short)0, acc[mt][g], false, false);

        if (wave == 0 && havenext)
            __builtin_amdgcn_s_wait_tensorcnt(0);   // DMA done (overlapped)
        __syncthreads();                             // publish buffer to block
    }

    // Epilogue: bias + gates + cell update, fused in-register.
    // C/D layout: lane col n; VGPR r -> row base + r + 8*lh
    const int n = by * BN + ni * 16 + lm;
    const float bI = bias[0 * HIDDEN + n];
    const float bF = bias[1 * HIDDEN + n];
    const float bG = bias[2 * HIDDEN + n];
    const float bO = bias[3 * HIDDEN + n];

#pragma unroll
    for (int mt = 0; mt < 4; ++mt) {
        const int mbase = bx * BM + mi * 64 + mt * 16;
#pragma unroll
        for (int r = 0; r < 8; ++r) {
            const int mr = mbase + r + lh * 8;
            const size_t idx = (size_t)mr * HIDDEN + n;
            float ig = sigmoidf_(acc[mt][0][r] + bI);
            float fg = sigmoidf_(acc[mt][1][r] + bF);
            float gg = tanhf(acc[mt][2][r] + bG);
            float og = sigmoidf_(acc[mt][3][r] + bO);
            float cn = fg * cst[idx] + ig * gg;
            cst[idx] = cn;
            float hn = og * tanhf(cn);
            hout_f32[idx] = hn;
            hout_bf[idx] = f2bf_rne(hn);
        }
    }
}

// ---------------------------------------------------------------------------
// Kernel 4: dense chain h@w2 -> tanh -> *w3 -> tanh -> *w4 -> tanh.
//   One wave per row of h[2048][1024]. Block = 8 waves, grid = 256.
//   Output order: real_pred [1024][20] then fake_pred [1024][20].
// ---------------------------------------------------------------------------
__global__ void __launch_bounds__(256)
k_dense_chain(const float* __restrict__ h,    // [2048][1024]
              const float* __restrict__ w2,   // [1024]
              const float* __restrict__ b2,
              const float* __restrict__ w3, const float* __restrict__ b3,
              const float* __restrict__ w4, const float* __restrict__ b4,
              float* __restrict__ out, int kidx)
{
    const int lane = threadIdx.x & 31;
    const int row = blockIdx.x * 8 + (threadIdx.x >> 5);
    const float* hr = h + (size_t)row * HIDDEN;

    float s = 0.f;
#pragma unroll 8
    for (int j = lane; j < HIDDEN; j += 32) s += hr[j] * w2[j];
#pragma unroll
    for (int off = 16; off; off >>= 1) s += __shfl_xor(s, off, 32);

    if (lane == 0) {
        float x = tanhf(s + b2[0]);
        x = tanhf(x * w3[0] + b3[0]);
        x = tanhf(x * w4[0] + b4[0]);
        int o = (row < BATCH) ? (row * PRED + kidx)
                              : (BATCH * PRED + (row - BATCH) * PRED + kidx);
        out[o] = x;
    }
}

// ---------------------------------------------------------------------------
// Host launcher
// ---------------------------------------------------------------------------
extern "C" void kernel_launch(void* const* d_in, const int* in_sizes, int n_in,
                              void* d_out, int out_size, void* d_ws, size_t ws_size,
                              hipStream_t stream) {
    const float* real_in = (const float*)d_in[0];   // [1024][30][512]
    const float* fake_in = (const float*)d_in[1];   // [1024][20][512]
    const float* Wx      = (const float*)d_in[2];   // [512][4096]
    const float* Wh      = (const float*)d_in[3];   // [1024][4096]
    const float* bias    = (const float*)d_in[4];   // [4096]
    const float* w2      = (const float*)d_in[5];   // [1024]
    const float* b2      = (const float*)d_in[6];
    const float* w3      = (const float*)d_in[7];
    const float* b3      = (const float*)d_in[8];
    const float* w4      = (const float*)d_in[9];
    const float* b4      = (const float*)d_in[10];
    float* out = (float*)d_out;

    // Workspace carve-up (256-byte aligned slices)
    char* ws = (char*)d_ws;
    size_t off = 0;
    auto take = [&](size_t bytes) -> void* {
        void* p = ws + off;
        off = (off + bytes + 255) & ~(size_t)255;
        return p;
    };
    unsigned short* wcat = (unsigned short*)take((size_t)GATES4 * KCAT * 2);         // 12.6 MB
    unsigned short* rbf  = (unsigned short*)take((size_t)TSTEPS * BATCH * FEAT * 2); // 31.5 MB
    unsigned short* fbf  = (unsigned short*)take((size_t)PRED * BATCH * FEAT * 2);   // 21.0 MB
    unsigned short* hbf0 = (unsigned short*)take((size_t)M2 * HIDDEN * 2);           // 4.2 MB
    unsigned short* hbf1 = (unsigned short*)take((size_t)M2 * HIDDEN * 2);           // 4.2 MB
    float* cbuf = (float*)take((size_t)M2 * HIDDEN * 4);                             // 8.4 MB
    float* hf32 = (float*)take((size_t)M2 * HIDDEN * 4);                             // 8.4 MB

    // Zero initial LSTM state (graph-capturable stream ops)
    (void)hipMemsetAsync(hbf0, 0, (size_t)M2 * HIDDEN * 2, stream);
    (void)hipMemsetAsync(cbuf, 0, (size_t)M2 * HIDDEN * 4, stream);

    // One-time precision conversion / layout transforms
    k_build_wcat<<<(GATES4 * KCAT) / 256, 256, 0, stream>>>(Wh, Wx, wcat);
    k_transpose_in<<<(TSTEPS * BATCH * FEAT) / 256, 256, 0, stream>>>(real_in, rbf, TSTEPS);
    k_transpose_in<<<(PRED * BATCH * FEAT) / 256, 256, 0, stream>>>(fake_in, fbf, PRED);

    unsigned short* hb[2] = {hbf0, hbf1};
    for (int t = 0; t < TSTEPS; ++t) {
        const unsigned short* xr = rbf + (size_t)t * BATCH * FEAT;
        const unsigned short* xf = (t < PREV)
            ? xr
            : (fbf + (size_t)(t - PREV) * BATCH * FEAT);
        k_lstm_step<<<dim3(M2 / BM, HIDDEN / BN), 256, 0, stream>>>(
            hb[t & 1], xr, xf, wcat, bias, cbuf, hb[(t + 1) & 1], hf32);
        if (t >= PREV) {
            k_dense_chain<<<M2 / 8, 256, 0, stream>>>(
                hf32, w2, b2, w3, b3, w4, b4, out, t - PREV);
        }
    }
    (void)in_sizes; (void)n_in; (void)out_size; (void)ws_size;
}
